// SwinTransformerBlock_9844065042699
// MI455X (gfx1250) — matrix operand order repaired and checked by reading, not compiled, not run on hardware
//
#include <hip/hip_runtime.h>

// ---------------------------------------------------------------------------
// Types (avoid HIP struct-vector types inside unions: use clang ext vectors)
// ---------------------------------------------------------------------------
typedef __bf16 bf16;
typedef __bf16        v16bf  __attribute__((ext_vector_type(16)));
typedef float         v8f    __attribute__((ext_vector_type(8)));
typedef unsigned int  u32x4  __attribute__((ext_vector_type(4)));
typedef float         f32x4  __attribute__((ext_vector_type(4)));

union Frag { u32x4 u[2]; v16bf v; };   // 32 bytes = one 16-bit WMMA operand

// Problem constants
#define BB   32
#define HH   64
#define WW   64
#define CC   256
#define WS   8
#define SHIFT 4
#define NHD  8
#define DD   32          // head dim
#define NN   64          // tokens per window
#define NWIN 64          // windows per image
#define HID  1024
#define MTOT (BB * NWIN * NN)   // 131072 rows

static __device__ __forceinline__ int regid(int p) {
    return p < (HH - WS) ? 0 : (p < (HH - SHIFT) ? 1 : 2);
}

// Low 32 bits of a generic pointer to __shared__ = LDS byte offset (ISA 10.2:
// LDS aperture addresses truncate to LDS_ADDR.U32).
static __device__ __forceinline__ unsigned int lds_off(const void* p) {
    return (unsigned int)(size_t)p;
}

// Async DMA: global -> LDS, 16 bytes per lane, tracked by ASYNCcnt.
static __device__ __forceinline__ void async_copy_b128(unsigned int ldsaddr,
                                                       const void* gaddr) {
    asm volatile("global_load_async_to_lds_b128 %0, %1, off"
                 :: "v"(ldsaddr), "v"((unsigned long long)(size_t)gaddr)
                 : "memory");
}
static __device__ __forceinline__ void async_copy_b32(unsigned int ldsaddr,
                                                      const void* gaddr) {
    asm volatile("global_load_async_to_lds_b32 %0, %1, off"
                 :: "v"(ldsaddr), "v"((unsigned long long)(size_t)gaddr)
                 : "memory");
}
static __device__ __forceinline__ void wait_async0() {
    asm volatile("s_wait_asynccnt 0" ::: "memory");
}

// ---------------------------------------------------------------------------
// f32 -> bf16 conversion (weights)
// ---------------------------------------------------------------------------
__global__ void swin_f32_to_bf16(const float* __restrict__ s, bf16* __restrict__ d, int n) {
    int i = blockIdx.x * 256 + threadIdx.x;
    if (i < n) d[i] = (bf16)s[i];
}

// ---------------------------------------------------------------------------
// LN1 + roll(-4,-4) + window partition -> bf16 rows in window-token order
// grid: B*nW blocks, 64 threads (one token each)
// ---------------------------------------------------------------------------
__global__ __launch_bounds__(64)
void swin_ln1_window(const float* __restrict__ x, const float* __restrict__ g,
                     const float* __restrict__ bta, bf16* __restrict__ y) {
    int blk = blockIdx.x;              // b*64 + widx
    int t   = threadIdx.x;             // token in window (r*8+c)
    int b   = blk >> 6, widx = blk & 63;
    int wh  = widx >> 3, ww = widx & 7;
    int r   = t >> 3,   c  = t & 7;
    int hh  = (wh * WS + r + SHIFT) & (HH - 1);
    int w2  = (ww * WS + c + SHIFT) & (WW - 1);
    const float* src = x + ((size_t)b * (HH * WW) + hh * WW + w2) * CC;

    float sum = 0.f, sq = 0.f;
    for (int i = 0; i < CC; i += 4) {
        f32x4 v = *(const f32x4*)(src + i);
        sum += v.x + v.y + v.z + v.w;
        sq  += v.x * v.x + v.y * v.y + v.z * v.z + v.w * v.w;
    }
    float m   = sum * (1.f / CC);
    float var = sq * (1.f / CC) - m * m;
    float inv = rsqrtf(var + 1e-5f);

    bf16* dst = y + ((size_t)blk * NN + t) * CC;
    for (int i = 0; i < CC; i++)
        dst[i] = (bf16)((src[i] - m) * inv * g[i] + bta[i]);
}

// ---------------------------------------------------------------------------
// LN2: z (natural token order, f32) -> bf16
// ---------------------------------------------------------------------------
__global__ __launch_bounds__(64)
void swin_ln2(const float* __restrict__ z, const float* __restrict__ g,
              const float* __restrict__ bta, bf16* __restrict__ y) {
    size_t tok = (size_t)blockIdx.x * 64 + threadIdx.x;
    const float* src = z + tok * CC;
    float sum = 0.f, sq = 0.f;
    for (int i = 0; i < CC; i += 4) {
        f32x4 v = *(const f32x4*)(src + i);
        sum += v.x + v.y + v.z + v.w;
        sq  += v.x * v.x + v.y * v.y + v.z * v.z + v.w * v.w;
    }
    float m   = sum * (1.f / CC);
    float inv = rsqrtf(sq * (1.f / CC) - m * m + 1e-5f);
    bf16* dst = y + tok * CC;
    for (int i = 0; i < CC; i++)
        dst[i] = (bf16)((src[i] - m) * inv * g[i] + bta[i]);
}

// ---------------------------------------------------------------------------
// Generic bf16 WMMA GEMM: out = epilogue(A[M,K] @ W[K,Nc] + bias)
//   128 threads = 4 waves; block tile 64x64; each wave 32x32 (2x2 WMMA tiles)
//   A tile staged by async global->LDS DMA (ASYNCcnt); B tile transposed
//   through VGPRs (DMA cannot transpose).
// Epilogues: 0 = qkv (bf16 out), 1 = proj (+x residual, window-reverse+roll,
//            f32 out), 2 = fc1 (GELU, bf16 out), 3 = fc2 (+z residual, f32 out)
// ---------------------------------------------------------------------------
template <int EPI>
__global__ __launch_bounds__(128)
void swin_gemm_bf16(const bf16* __restrict__ A, const bf16* __restrict__ W,
                    const float* __restrict__ bias, const float* __restrict__ extra,
                    void* __restrict__ outp, int M, int K, int Nc) {
    __shared__ __align__(16) bf16 Asm[64 * 40];   // A tile, row-major, pad 40
    __shared__ __align__(16) bf16 Bts[64 * 40];   // B tile transposed [n][k]

    const int tid = threadIdx.x, lane = tid & 31, wv = tid >> 5;
    const int half = lane >> 4, ln16 = lane & 15;
    const int mBlocks = M >> 6;
    const int bm0 = (blockIdx.x % mBlocks) << 6;
    const int bn0 = (blockIdx.x / mBlocks) << 6;
    const int wr = wv >> 1, wc = wv & 1;

    v8f zero = {};
    v8f acc[2][2];
    acc[0][0] = zero; acc[0][1] = zero; acc[1][0] = zero; acc[1][1] = zero;

    const int nkb = K >> 5;
    for (int kb = 0; kb < nkb; kb++) {
        // ---- stage A tile [64 x 32] via async DMA (2 x b128 per lane) ----
        #pragma unroll
        for (int it = 0; it < 2; it++) {
            int ci = tid + it * 128;            // 256 x 16B chunks
            int r = ci >> 2, cs = ci & 3;
            async_copy_b128(lds_off(&Asm[r * 40 + cs * 8]),
                            A + (size_t)(bm0 + r) * K + kb * 32 + cs * 8);
        }
        // ---- stage Bt tile: Bt[n][k] = W[kb*32+k][bn0+n] (transpose) ----
        #pragma unroll
        for (int it = 0; it < 2; it++) {
            int ci = tid + it * 128;
            int k = ci >> 3, ns = ci & 7;
            u32x4 v = *(const u32x4*)(W + (size_t)(kb * 32 + k) * Nc + bn0 + ns * 8);
            const bf16* pv = (const bf16*)&v;
            #pragma unroll
            for (int j = 0; j < 8; j++) Bts[(ns * 8 + j) * 40 + k] = pv[j];
        }
        if (kb + 1 < nkb) {   // emits global_prefetch_b8
            __builtin_prefetch(A + (size_t)(bm0 + (tid >> 1)) * K + (kb + 1) * 32 + (tid & 1) * 16, 0, 1);
            __builtin_prefetch(W + (size_t)((kb + 1) * 32 + (tid >> 2)) * Nc + bn0 + (tid & 3) * 16, 0, 1);
        }
        wait_async0();                           // A-tile DMA complete
        __syncthreads();

        Frag a[2], bfr[2];
        #pragma unroll
        for (int mi = 0; mi < 2; mi++) {
            int row = wr * 32 + mi * 16 + ln16;
            a[mi].u[0] = *(const u32x4*)(&Asm[row * 40 + 8 * half]);
            a[mi].u[1] = *(const u32x4*)(&Asm[row * 40 + 16 + 8 * half]);
        }
        #pragma unroll
        for (int ni = 0; ni < 2; ni++) {
            int n = wc * 32 + ni * 16 + ln16;
            bfr[ni].u[0] = *(const u32x4*)(&Bts[n * 40 + 8 * half]);
            bfr[ni].u[1] = *(const u32x4*)(&Bts[n * 40 + 16 + 8 * half]);
        }
        #pragma unroll
        for (int mi = 0; mi < 2; mi++)
            #pragma unroll
            for (int ni = 0; ni < 2; ni++)
                acc[mi][ni] = __builtin_amdgcn_wmma_f32_16x16x32_bf16(
                    false, a[mi].v, false, bfr[ni].v, (short)0, acc[mi][ni], false, false);
        __syncthreads();
    }

    // ---- epilogue ----
    #pragma unroll
    for (int mi = 0; mi < 2; mi++) {
        #pragma unroll
        for (int ni = 0; ni < 2; ni++) {
            #pragma unroll
            for (int i = 0; i < 8; i++) {
                int row = bm0 + wr * 32 + mi * 16 + i + 8 * half;
                int col = bn0 + wc * 32 + ni * 16 + ln16;
                float v = acc[mi][ni][i] + bias[col];
                if (EPI == 0) {                        // qkv -> bf16
                    ((bf16*)outp)[(size_t)row * Nc + col] = (bf16)v;
                } else if (EPI == 1) {                 // proj: window reverse + roll + residual
                    int b = row >> 12, widx = (row >> 6) & 63, t = row & 63;
                    int wh = widx >> 3, wwc = widx & 7, r = t >> 3, c = t & 7;
                    int hh = (wh * WS + r + SHIFT) & (HH - 1);
                    int w2 = (wwc * WS + c + SHIFT) & (WW - 1);
                    size_t tgt = ((size_t)b * (HH * WW) + hh * WW + w2) * CC + col;
                    ((float*)outp)[tgt] = v + extra[tgt];
                } else if (EPI == 2) {                 // fc1: exact GELU -> bf16
                    float ge = 0.5f * v * (1.0f + erff(v * 0.70710678118654752f));
                    ((bf16*)outp)[(size_t)row * Nc + col] = (bf16)ge;
                } else {                               // fc2: + z -> f32 out
                    size_t idx = (size_t)row * Nc + col;
                    ((float*)outp)[idx] = v + extra[idx];
                }
            }
        }
    }
}

// ---------------------------------------------------------------------------
// Attention: one block per (window, head). 128 threads = 4 waves.
//   S = scale * q k^T + rpb + shift-mask ; softmax ; O = P V
// ---------------------------------------------------------------------------
__global__ __launch_bounds__(128)
void swin_attn(const bf16* __restrict__ qkv, const float* __restrict__ rpb_table,
               bf16* __restrict__ o) {
    __shared__ __align__(16) float S[64 * 68];
    __shared__ __align__(16) bf16  P[64 * 72];
    __shared__ __align__(16) bf16  Vt[32 * 72];
    __shared__ __align__(16) float rpbs[(2 * WS - 1) * (2 * WS - 1) * NHD];   // 1800

    const int tid = threadIdx.x, lane = tid & 31, wv = tid >> 5;
    const int half = lane >> 4, ln16 = lane & 15;
    const int win = blockIdx.x >> 3, hd = blockIdx.x & 7;
    const bf16* qb = qkv + (size_t)win * NN * (3 * CC) + hd * DD;
    const bf16* kbp = qb + CC;
    const bf16* vbp = qb + 2 * CC;

    // rpb table -> LDS via async DMA (ASYNCcnt)
    for (int i = tid; i < 1800; i += 128)
        async_copy_b32(lds_off(&rpbs[i]), rpb_table + i);

    // stage V^T : Vt[d][tok] (transpose, through VGPRs)
    #pragma unroll
    for (int e = 0; e < 16; e++) {
        int idx = tid * 16 + e;
        int tok = idx >> 5, d = idx & 31;
        Vt[d * 72 + tok] = vbp[(size_t)tok * (3 * CC) + d];
    }
    wait_async0();
    __syncthreads();     // rpbs/Vt visible to all threads before use

    // ---- S = scale*q@k^T + rpb + mask, wave wv owns rows [16wv,16wv+16) ----
    const int m0 = wv * 16;
    Frag fa;
    {
        const bf16* qr = qb + (size_t)(m0 + ln16) * (3 * CC);
        fa.u[0] = *(const u32x4*)(qr + 8 * half);
        fa.u[1] = *(const u32x4*)(qr + 16 + 8 * half);
    }
    const int widx = win & 63, wh = widx >> 3, wwc = widx & 7;
    const float scale = 0.17677669529663687f;   // 32^-0.5

    #pragma unroll
    for (int j = 0; j < 4; j++) {
        Frag fb;
        const int jt = j * 16 + ln16;           // key token for this lane's column
        const bf16* kr = kbp + (size_t)jt * (3 * CC);
        fb.u[0] = *(const u32x4*)(kr + 8 * half);
        fb.u[1] = *(const u32x4*)(kr + 16 + 8 * half);
        v8f accS = {};
        accS = __builtin_amdgcn_wmma_f32_16x16x32_bf16(
            false, fa.v, false, fb.v, (short)0, accS, false, false);

        int rj = jt >> 3, cj = jt & 7;
        int idj = 3 * regid(wh * WS + rj) + regid(wwc * WS + cj);
        #pragma unroll
        for (int i = 0; i < 8; i++) {
            int it = m0 + i + 8 * half;
            int ri = it >> 3, ci = it & 7;
            int rel = (ri - rj + WS - 1) * (2 * WS - 1) + (ci - cj + WS - 1);
            int idi = 3 * regid(wh * WS + ri) + regid(wwc * WS + ci);
            float v = scale * accS[i] + rpbs[rel * NHD + hd] + ((idi != idj) ? -100.f : 0.f);
            S[it * 68 + jt] = v;
        }
    }
    __syncthreads();

    // ---- softmax, one row per thread (rows 0..63) ----
    if (tid < 64) {
        float mx = -1e30f;
        for (int j = 0; j < NN; j++) mx = fmaxf(mx, S[tid * 68 + j]);
        float sum = 0.f;
        for (int j = 0; j < NN; j++) sum += __expf(S[tid * 68 + j] - mx);
        float inv = 1.f / sum;
        for (int j = 0; j < NN; j++)
            P[tid * 72 + j] = (bf16)(__expf(S[tid * 68 + j] - mx) * inv);
    }
    __syncthreads();

    // ---- O = P @ V : wave wv owns rows [16wv,16wv+16), two 16-wide d tiles ----
    #pragma unroll
    for (int nt = 0; nt < 2; nt++) {
        v8f accO = {};
        #pragma unroll
        for (int kb2 = 0; kb2 < 2; kb2++) {
            Frag pa, pb;
            const char* pr = (const char*)&P[(m0 + ln16) * 72];
            pa.u[0] = *(const u32x4*)(pr + kb2 * 64 + 16 * half);
            pa.u[1] = *(const u32x4*)(pr + kb2 * 64 + 32 + 16 * half);
            const char* vr = (const char*)&Vt[(nt * 16 + ln16) * 72];
            pb.u[0] = *(const u32x4*)(vr + kb2 * 64 + 16 * half);
            pb.u[1] = *(const u32x4*)(vr + kb2 * 64 + 32 + 16 * half);
            accO = __builtin_amdgcn_wmma_f32_16x16x32_bf16(
                false, pa.v, false, pb.v, (short)0, accO, false, false);
        }
        #pragma unroll
        for (int i = 0; i < 8; i++) {
            int tok = m0 + i + 8 * half;
            int dcol = nt * 16 + ln16;
            o[((size_t)win * NN + tok) * CC + hd * DD + dcol] = (bf16)accO[i];
        }
    }
}

// ---------------------------------------------------------------------------
// Launch
// ---------------------------------------------------------------------------
extern "C" void kernel_launch(void* const* d_in, const int* in_sizes, int n_in,
                              void* d_out, int out_size, void* d_ws, size_t ws_size,
                              hipStream_t stream) {
    const float* x      = (const float*)d_in[0];
    const float* ln1_g  = (const float*)d_in[1];
    const float* ln1_b  = (const float*)d_in[2];
    const float* qkv_w  = (const float*)d_in[3];
    const float* qkv_b  = (const float*)d_in[4];
    const float* rpbt   = (const float*)d_in[5];
    const float* proj_w = (const float*)d_in[6];
    const float* proj_b = (const float*)d_in[7];
    const float* ln2_g  = (const float*)d_in[8];
    const float* ln2_b  = (const float*)d_in[9];
    const float* fc1_w  = (const float*)d_in[10];
    const float* fc1_b  = (const float*)d_in[11];
    const float* fc2_w  = (const float*)d_in[12];
    const float* fc2_b  = (const float*)d_in[13];
    float* out = (float*)d_out;

    // workspace layout (bytes)
    char* ws = (char*)d_ws;
    bf16* wq  = (bf16*)ws;                      // 256*768
    bf16* wp  = wq + 256 * 768;                 // 256*256
    bf16* w1  = wp + 256 * 256;                 // 256*1024
    bf16* w2  = w1 + 256 * 1024;                // 1024*256
    char* p   = (char*)(w2 + 1024 * 256);       // 1,572,864 B so far
    bf16* ybuf    = (bf16*)p;                   // 64 MiB (reused as ln2 output)
    p += (size_t)MTOT * CC * sizeof(bf16);
    bf16* qkvbuf  = (bf16*)p;                   // 192 MiB
    bf16* attnout = (bf16*)(p + (size_t)MTOT * 3 * CC * sizeof(bf16));  // 64 MiB
    bf16* h1      = (bf16*)p;                   // 256 MiB, reuses qkv+attnout region
    p += (size_t)MTOT * HID * sizeof(bf16);
    float* zbuf   = (float*)p;                  // 128 MiB

    // 1) weights -> bf16
    swin_f32_to_bf16<<<(256 * 768 + 255) / 256, 256, 0, stream>>>(qkv_w, wq, 256 * 768);
    swin_f32_to_bf16<<<(256 * 256 + 255) / 256, 256, 0, stream>>>(proj_w, wp, 256 * 256);
    swin_f32_to_bf16<<<(256 * 1024 + 255) / 256, 256, 0, stream>>>(fc1_w, w1, 256 * 1024);
    swin_f32_to_bf16<<<(1024 * 256 + 255) / 256, 256, 0, stream>>>(fc2_w, w2, 1024 * 256);

    // 2) LN1 + shift + window partition
    swin_ln1_window<<<BB * NWIN, 64, 0, stream>>>(x, ln1_g, ln1_b, ybuf);

    // 3) qkv GEMM  [131072,256] x [256,768]
    swin_gemm_bf16<0><<<(MTOT / 64) * (768 / 64), 128, 0, stream>>>(
        ybuf, wq, qkv_b, nullptr, qkvbuf, MTOT, CC, 3 * CC);

    // 4) windowed attention, one block per (window, head)
    swin_attn<<<BB * NWIN * NHD, 128, 0, stream>>>(qkvbuf, rpbt, attnout);

    // 5) proj GEMM + window reverse + roll + residual -> z (f32)
    swin_gemm_bf16<1><<<(MTOT / 64) * (CC / 64), 128, 0, stream>>>(
        attnout, wp, proj_b, x, zbuf, MTOT, CC, CC);

    // 6) LN2
    swin_ln2<<<MTOT / 64, 64, 0, stream>>>(zbuf, ln2_g, ln2_b, ybuf);

    // 7) fc1 GEMM + GELU
    swin_gemm_bf16<2><<<(MTOT / 64) * (HID / 64), 128, 0, stream>>>(
        ybuf, w1, fc1_b, nullptr, h1, MTOT, CC, HID);

    // 8) fc2 GEMM + residual -> out (f32)
    swin_gemm_bf16<3><<<(MTOT / 64) * (CC / 64), 128, 0, stream>>>(
        h1, w2, fc2_b, zbuf, out, MTOT, HID, CC);
}
